// SpikingParrot_6871947674304
// MI455X (gfx1250) — compile-verified, matrix operand order checked by reading, not run on hardware
//
#include <hip/hip_runtime.h>
#include <hip/hip_bf16.h>

// ======================= types / WMMA helpers =======================
typedef __bf16 bf16_t;
typedef __attribute__((ext_vector_type(16))) __bf16 v16bf;
typedef __attribute__((ext_vector_type(8)))  __bf16 v8bf;
typedef __attribute__((ext_vector_type(8)))  float  v8f;

__device__ __forceinline__ v8f wmma_bf16(v16bf a, v16bf b, v8f c) {
  // D = A(16x32 bf16) * B(32x16 bf16) + C(16x16 f32)
  return __builtin_amdgcn_wmma_f32_16x16x32_bf16(
      /*neg_a=*/false, a, /*neg_b=*/false, b,
      /*c_mod=*/(short)0, c, /*reuse_a=*/false, /*reuse_b=*/false);
}

// Load a 16x32 (or 32x16, symmetric) bf16 operand fragment.
// Per ISA 7.12.2: lane = {half,row}; per-lane 16 values are
// K = k0 + half*8 + 0..7  and  K = k0 + 16 + half*8 + 0..7  -> two b128 loads.
__device__ __forceinline__ v16bf load_frag(const bf16_t* base, int ld, int k0) {
  int lane = threadIdx.x & 31;
  int row  = lane & 15;
  int half = lane >> 4;
  const bf16_t* p = base + (long)row * ld + k0 + half * 8;
  union { v16bf v; v8bf h[2]; } u;
  u.h[0] = *(const v8bf*)p;
  u.h[1] = *(const v8bf*)(p + 16);
  return u.v;
}

// ---- CDNA5 async global->LDS staging (ASYNCcnt path), guarded ----
#if __has_builtin(__builtin_amdgcn_global_load_async_to_lds_b128)
#define HAVE_ASYNC_LDS 1
#else
#define HAVE_ASYNC_LDS 0
#endif

template <int N>
__device__ __forceinline__ void wait_async_le() {
#if __has_builtin(__builtin_amdgcn_s_wait_asynccnt)
  __builtin_amdgcn_s_wait_asynccnt(N);
#else
  asm volatile("s_wait_asynccnt %0" :: "i"(N) : "memory");
#endif
}

// 16-byte global -> LDS stage (async when available, else load+store).
__device__ __forceinline__ void stage_b128(const bf16_t* g, bf16_t* l) {
#if HAVE_ASYNC_LDS
  typedef int v4i_vec __attribute__((ext_vector_type(4)));
  typedef __attribute__((address_space(1))) v4i_vec* as1_v4i_p;
  typedef __attribute__((address_space(3))) v4i_vec* as3_v4i_p;
  __builtin_amdgcn_global_load_async_to_lds_b128(
      (as1_v4i_p)(unsigned long long)g,
      (as3_v4i_p)(unsigned)(unsigned long long)l,  // flat->LDS offset truncation
      0, 0);
#else
  *(v8bf*)l = *(const v8bf*)g;
#endif
}

// ---- TDM: contiguous copy of n bf16 elements into LDS via tensor DMA ----
// D# per ISA 8.3/8.4: group0 = {count=1 | lds_addr | global_addr | type=2},
// group1 = {data_size=2B, tensor_dim0=n, tensor_dim1=1, tile_dim0=n,
//           tile_dim1=1, stride0=n}. Groups 2/3 + trailing vector zero (<=2D).
__device__ __forceinline__ void tdm_copy_to_lds(const void* gsrc, void* lds_dst, int n) {
#if __has_builtin(__builtin_amdgcn_tensor_load_to_lds)
  typedef __attribute__((ext_vector_type(4))) unsigned v4u_t;
  typedef __attribute__((ext_vector_type(8))) int      v8i_t;
  typedef __attribute__((ext_vector_type(4))) int      v4i_t;
  unsigned long long ga = (unsigned long long)gsrc;
  unsigned lds_off = (unsigned)(unsigned long long)lds_dst;  // flat->LDS truncation
  v4u_t g0;
  g0[0] = 1u;                                   // count=1, user mode
  g0[1] = lds_off;                              // lds_addr
  g0[2] = (unsigned)ga;                         // global_addr[31:0]
  g0[3] = (unsigned)((ga >> 32) & 0x01FFFFFFull) | (2u << 30);  // addr[56:32]|type=2
  v8i_t g1;
  g1[0] = 0x10000;                              // data_size=1 (2 bytes), wg_mask=0
  g1[1] = (n & 0xFFFF) << 16;                   // tensor_dim0[15:0] @bit48
  g1[2] = ((n >> 16) & 0xFFFF) | (1 << 16);     // tensor_dim0[31:16], tensor_dim1=1
  g1[3] = (n & 0xFFFF) << 16;                   // tile_dim0 @bit112
  g1[4] = 1;                                    // tile_dim1=1, tile_dim2=0
  g1[5] = n;                                    // tensor_dim0_stride[31:0]
  g1[6] = 0;
  g1[7] = 0;
  v4i_t z4 = {0, 0, 0, 0};
  v8i_t z8 = {0, 0, 0, 0, 0, 0, 0, 0};
  __builtin_amdgcn_tensor_load_to_lds(g0, g1, z4, z4, z8, 0);
#else
  (void)gsrc; (void)lds_dst; (void)n;
#endif
}

// ======================= small utility kernels =======================
__global__ void k_zero_u32(unsigned* p) { *p = 0u; }

__global__ void k_cvt(const float* __restrict__ s, bf16_t* __restrict__ d, long n) {
  long i = ((long)blockIdx.x * blockDim.x + threadIdx.x) * 4;
  if (i + 3 < n) {
    float4 v = *(const float4*)(s + i);
    d[i + 0] = (bf16_t)v.x; d[i + 1] = (bf16_t)v.y;
    d[i + 2] = (bf16_t)v.z; d[i + 3] = (bf16_t)v.w;
  } else {
    for (; i < n; ++i) d[i] = (bf16_t)s[i];
  }
}

// Embedding gather + DyT norm -> bf16; optional time-flipped copy (backward enc).
__global__ void k_embed_dyt(const int* __restrict__ tok, const float* __restrict__ emb,
                            const float* __restrict__ alpha_p,
                            const float* __restrict__ gamma, const float* __restrict__ beta,
                            bf16_t* __restrict__ out, bf16_t* __restrict__ out_flip,
                            int T, int E) {
  int bt = blockIdx.x;
  int b = bt / T, t = bt % T;
  long id = tok[bt];
  float al = alpha_p[0];
  for (int e = threadIdx.x; e < E; e += blockDim.x) {
    float v = emb[id * E + e];
    bf16_t r = (bf16_t)(gamma[e] * tanhf(al * v) + beta[e]);
    out[(long)bt * E + e] = r;
    if (out_flip) out_flip[((long)b * T + (T - 1 - t)) * E + e] = r;
  }
}

// ======================= batched WMMA GEMM (pipelined) =======================
// C[M,N] = A[M,K](bf16, row-major) * W[N,K]^T(bf16, row-major) + biases
// MODE 0: Cf[m*N+n] = acc + bias0 + bias1                           (fp32 out)
// MODE 1: Cb[m*N+n] = bf16( gamma*tanh(alpha*(acc+bias0)) + beta )  (DyT)
// Block tile 128x128, 256 threads = 8 waves (2x4); each wave 64x32:
// 4 A-frags x 2 B-frags -> 8 WMMA per k-step (A reused across both B tiles).
// Double-buffered LDS; async loads for step k+1 overlap WMMA on step k
// (async loads retire in issue order, so s_wait_asynccnt 4 releases the
// current panel while the 4 next-panel loads stay in flight).
// Requires M%128==0, N%128==0, K%32==0 (true for every call here).
template <int MODE>
__global__ void __launch_bounds__(256)
k_gemm(const bf16_t* __restrict__ A, int K,
       const bf16_t* __restrict__ W, int N,
       const float* __restrict__ bias0, const float* __restrict__ bias1,
       const float* __restrict__ alpha_p,
       const float* __restrict__ gamma, const float* __restrict__ beta,
       float* __restrict__ Cf, bf16_t* __restrict__ Cb) {
  const int LS = 48;  // LDS row stride (elems); 96B keeps 16B alignment, skews banks
  __shared__ __align__(16) bf16_t sA[2][128 * 48];
  __shared__ __align__(16) bf16_t sB[2][128 * 48];

  int m0 = blockIdx.y * 128;
  int n0 = blockIdx.x * 128;
  int tid = threadIdx.x;
  int wave = tid >> 5, lane = tid & 31;
  int wm = wave & 1, wn = wave >> 1;  // 2 x 4 wave grid; wave tile 64x32

  // stage panels for k-step k0 into buffer `buf`: 4 async b128 per thread
  auto stage = [&](int buf, int k0) {
    #pragma unroll
    for (int rep = 0; rep < 2; ++rep) {
      int idx = tid + rep * 256;
      int r = idx >> 2, c = (idx & 3) * 8;
      stage_b128(&A[(long)(m0 + r) * K + k0 + c], &sA[buf][r * LS + c]);
      stage_b128(&W[(long)(n0 + r) * K + k0 + c], &sB[buf][r * LS + c]);
    }
  };

  const int NK = K >> 5;
  stage(0, 0);
  v8f acc[4][2] = {};
  for (int kt = 0; kt < NK; ++kt) {
    __syncthreads();  // everyone done reading the buffer about to be restaged
    if (kt + 1 < NK) {
      stage((kt + 1) & 1, (kt + 1) << 5);
      wait_async_le<4>();  // current panel complete; next panel in flight
    } else {
      wait_async_le<0>();
    }
    __syncthreads();
    const bf16_t* cA = sA[kt & 1];
    const bf16_t* cB = sB[kt & 1];
    v16bf af[4], bfr[2];
    #pragma unroll
    for (int i = 0; i < 4; ++i) af[i] = load_frag(&cA[(wm * 64 + i * 16) * LS], LS, 0);
    #pragma unroll
    for (int j = 0; j < 2; ++j) bfr[j] = load_frag(&cB[(wn * 32 + j * 16) * LS], LS, 0);
    #pragma unroll
    for (int i = 0; i < 4; ++i)
      #pragma unroll
      for (int j = 0; j < 2; ++j)
        acc[i][j] = wmma_bf16(af[i], bfr[j], acc[i][j]);
  }

  // Epilogue. D layout: VGPR r, lane l -> m = r + 8*(l>>4), n = l&15.
  int lhalf = lane >> 4, lcol = lane & 15;
  float al = (MODE == 1) ? alpha_p[0] : 0.f;
  #pragma unroll
  for (int i = 0; i < 4; ++i)
    #pragma unroll
    for (int j = 0; j < 2; ++j) {
      int mbase = m0 + wm * 64 + i * 16 + 8 * lhalf;
      int n = n0 + wn * 32 + j * 16 + lcol;
      float b0 = bias0 ? bias0[n] : 0.f;
      float b1 = bias1 ? bias1[n] : 0.f;
      #pragma unroll
      for (int r = 0; r < 8; ++r) {
        long m = mbase + r;
        float v = acc[i][j][r] + b0 + b1;
        if (MODE == 0) Cf[m * N + n] = v;
        else           Cb[m * N + n] = (bf16_t)(gamma[n] * tanhf(al * v) + beta[n]);
      }
    }
}

// ======================= persistent SLSTM scan =======================
// Monotonic global barrier: safe across repeated calls once counter is zeroed.
__device__ __forceinline__ void grid_barrier(unsigned* cnt, unsigned nwg) {
  __syncthreads();
  if (threadIdx.x == 0) {
    unsigned old = atomicAdd(cnt, 1u);
    unsigned target = (old / nwg + 1u) * nwg;
    while (__hip_atomic_load(cnt, __ATOMIC_ACQUIRE, __HIP_MEMORY_SCOPE_AGENT) < target)
      __builtin_amdgcn_s_sleep(1);
  }
  __syncthreads();
}

// Per timestep: gates = gates_x (precomputed, biases folded) + mem(t-1)·W_hh^T.
// A = mem as 16xH bf16 (rows 8..15 zero pad), ping-pong global buffer, staged
// into LDS each step via the Tensor Data Mover (fallback: vector copies).
// gridDim.x = H/128 workgroups; each wave owns one 16-wide hidden tile, computes
// all 4 gates for it (A-fragment reused 4x). syn/mem live in registers for all T.
__global__ void __launch_bounds__(256)
k_lstm_scan(const float* __restrict__ gatesx,   // [B*T, 4H], row = b*T + t
            const bf16_t* __restrict__ Whh,     // [4H, H]
            const float* __restrict__ thr_p,
            const float* __restrict__ init_syn, // [B, H] or null (zeros)
            const float* __restrict__ init_mem, // [B, H] or null
            bf16_t* __restrict__ spk,           // [B*T, H]
            float* __restrict__ out_syn, float* __restrict__ out_mem,
            int out_ld, int out_off,
            bf16_t* __restrict__ membuf,        // [2][16*H] ping-pong
            unsigned* __restrict__ bar,
            int H, int T) {
  extern __shared__ __align__(16) bf16_t sMem[];  // [16 * H]
  const int G = 4 * H;
  const unsigned nwg = gridDim.x;
  int tid = threadIdx.x, wave = tid >> 5, lane = tid & 31;
  int lhalf = lane >> 4, lcol = lane & 15;
  int n0 = blockIdx.x * 128 + wave * 16;  // this wave's hidden tile base
  int j = n0 + lcol;                      // hidden column (valid when lhalf==0)
  float thr = thr_p[0];

  bf16_t* buf0 = membuf;
  bf16_t* buf1 = membuf + (long)16 * H;

  // Init this block's 128-column slice of both ping-pong buffers.
  for (int idx = tid; idx < 16 * 128; idx += blockDim.x) {
    int r = idx >> 7;
    int c = blockIdx.x * 128 + (idx & 127);
    float v0 = (r < 8 && init_mem) ? init_mem[(long)r * H + c] : 0.f;
    buf0[(long)r * H + c] = (bf16_t)((r < 8) ? v0 : 0.f);
    buf1[(long)r * H + c] = (bf16_t)0.f;
  }
  float syn[8], mem[8];
  #pragma unroll
  for (int r = 0; r < 8; ++r) {
    syn[r] = init_syn ? init_syn[(long)r * H + j] : 0.f;
    mem[r] = init_mem ? init_mem[(long)r * H + j] : 0.f;
  }
  __threadfence();
  grid_barrier(bar, nwg);

  const int KT = H >> 5;
  for (int t = 0; t < T; ++t) {
    const bf16_t* rbuf = (t & 1) ? buf1 : buf0;
    bf16_t* wbuf = (t & 1) ? buf0 : buf1;

    __syncthreads();  // previous step's LDS consumers done
#if __has_builtin(__builtin_amdgcn_tensor_load_to_lds)
    if (tid < 32) {   // one wave issues the tensor DMA for the whole block
      tdm_copy_to_lds(rbuf, sMem, 16 * H);
      __builtin_amdgcn_s_wait_tensorcnt(0);
    }
    __syncthreads();
#else
    for (int idx = tid * 8; idx < 16 * H; idx += blockDim.x * 8)
      *(v8bf*)&sMem[idx] = *(const v8bf*)&rbuf[idx];
    __syncthreads();
#endif

    v8f acc[4] = {};
    for (int kt = 0; kt < KT; ++kt) {
      int k0 = kt * 32;
      v16bf a = load_frag(sMem, H, k0);  // shared across 4 gates
      #pragma unroll
      for (int g = 0; g < 4; ++g) {
        v16bf w = load_frag(Whh + (long)(g * H + n0) * H, H, k0);
        acc[g] = wmma_bf16(a, w, acc[g]);
      }
    }

    if (lhalf == 0) {  // rows m = 0..7 are the real batch
      #pragma unroll
      for (int r = 0; r < 8; ++r) {
        long row = (long)(r * T + t) * G;
        float xi = gatesx[row + 0 * H + j] + acc[0][r];
        float xf = gatesx[row + 1 * H + j] + acc[1][r];
        float xg = gatesx[row + 2 * H + j] + acc[2][r];
        float xo = gatesx[row + 3 * H + j] + acc[3][r];
        float ig = 1.f / (1.f + expf(-xi));
        float fg = 1.f / (1.f + expf(-xf));
        float gg = tanhf(xg);
        float og = 1.f / (1.f + expf(-xo));
        syn[r] = fg * syn[r] + ig * gg;
        mem[r] = og * tanhf(syn[r]);
        spk[(long)(r * T + t) * H + j] = (bf16_t)((mem[r] - thr) > 0.f ? 1.f : 0.f);
        wbuf[(long)r * H + j] = (bf16_t)mem[r];
      }
    }
    __threadfence();
    grid_barrier(bar, nwg);
  }

  if (lhalf == 0) {
    #pragma unroll
    for (int r = 0; r < 8; ++r) {
      out_syn[(long)r * out_ld + out_off + j] = syn[r];
      out_mem[(long)r * out_ld + out_off + j] = mem[r];
    }
  }
}

// ======================= host orchestration =======================
struct LayerP {
  const float *W_hh, *W_ih, *b_hh, *b_ih, *fc_W, *fc_b, *alpha, *beta, *gamma, *thr;
};
static inline LayerP layer_at(void* const* d_in, int base) {
  LayerP p;
  p.W_hh  = (const float*)d_in[base + 0];
  p.W_ih  = (const float*)d_in[base + 1];
  p.b_hh  = (const float*)d_in[base + 2];
  p.b_ih  = (const float*)d_in[base + 3];
  p.fc_W  = (const float*)d_in[base + 4];
  p.fc_b  = (const float*)d_in[base + 5];
  p.alpha = (const float*)d_in[base + 6];
  p.beta  = (const float*)d_in[base + 7];
  p.gamma = (const float*)d_in[base + 8];
  p.thr   = (const float*)d_in[base + 9];
  return p;
}

extern "C" void kernel_launch(void* const* d_in, const int* in_sizes, int n_in,
                              void* d_out, int out_size, void* d_ws, size_t ws_size,
                              hipStream_t stream) {
  (void)in_sizes; (void)out_size;
  constexpr int B = 8, T = 256, BT = 2048, E = 512, H = 512, HD = 1024, V = 32000;
  if (n_in < 68) return;  // expects JAX pytree (sorted-key) flattening: 68 leaves

  const int* src = (const int*)d_in[0];
  const int* tgt = (const int*)d_in[1];
  // sorted top-level keys: dec < embedding < enc_b < enc_f < fc_W < fc_b < norm
  LayerP dec0  = layer_at(d_in, 2);
  LayerP dec1p = layer_at(d_in, 12);
  const float* emb = (const float*)d_in[22];
  LayerP encb[2] = { layer_at(d_in, 23), layer_at(d_in, 33) };
  LayerP encf[2] = { layer_at(d_in, 43), layer_at(d_in, 53) };
  const float* fcW    = (const float*)d_in[63];
  const float* fcb    = (const float*)d_in[64];
  const float* nalpha = (const float*)d_in[65];
  const float* nbeta  = (const float*)d_in[66];
  const float* ngamma = (const float*)d_in[67];

  // ---- workspace carve (bump allocator) ----
  char* wsb = (char*)d_ws;
  size_t off = 0;
  auto alloc = [&](size_t bytes) -> void* {
    void* p = wsb + off;
    off = (off + bytes + 255) & ~(size_t)255;
    return p;
  };
  bf16_t* actA  = (bf16_t*)alloc((size_t)BT * E * 2);      // enc fwd ping
  bf16_t* actB  = (bf16_t*)alloc((size_t)BT * E * 2);      // enc fwd pong
  bf16_t* actC  = (bf16_t*)alloc((size_t)BT * E * 2);      // enc bwd ping (flipped)
  bf16_t* actD  = (bf16_t*)alloc((size_t)BT * E * 2);      // enc bwd pong
  bf16_t* tgtbuf= (bf16_t*)alloc((size_t)BT * E * 2);
  bf16_t* decA  = (bf16_t*)alloc((size_t)BT * HD * 2);
  bf16_t* decB  = (bf16_t*)alloc((size_t)BT * HD * 2);
  bf16_t* spkbuf= (bf16_t*)alloc((size_t)BT * HD * 2);
  float*  gates = (float*) alloc((size_t)BT * 4 * HD * 4); // 32 MB, reused per scan
  bf16_t* wih   = (bf16_t*)alloc((size_t)4096 * 1024 * 2); // W_ih / fc_W staging
  bf16_t* whh   = (bf16_t*)alloc((size_t)4096 * 1024 * 2);
  bf16_t* wbig  = (bf16_t*)alloc((size_t)V * HD * 2);      // final vocab proj
  float*  ssyn  = (float*) alloc((size_t)B * HD * 4);      // scratch finals
  float*  smem  = (float*) alloc((size_t)B * HD * 4);
  float*  dsyn  = (float*) alloc((size_t)B * HD * 4);      // dec layer0 init (concat f|b)
  float*  dmem  = (float*) alloc((size_t)B * HD * 4);
  bf16_t* membuf= (bf16_t*)alloc((size_t)2 * 16 * HD * 2);
  unsigned* bar = (unsigned*)alloc(256);
  if (off > ws_size) return;  // insufficient scratch

  auto cvt = [&](const float* s, bf16_t* d, long n) {
    unsigned blocks = (unsigned)((n / 4 + 255) / 256);
    k_cvt<<<blocks, 256, 0, stream>>>(s, d, n);
  };
  auto gemm_f32 = [&](const bf16_t* A, int K, const bf16_t* W, int N,
                      const float* b0, const float* b1, float* C) {
    dim3 g(N / 128, BT / 128);
    k_gemm<0><<<g, 256, 0, stream>>>(A, K, W, N, b0, b1, nullptr, nullptr, nullptr, C, nullptr);
  };
  auto gemm_dyt = [&](const bf16_t* A, int K, const bf16_t* W, int N,
                      const float* b0, const float* al, const float* ga, const float* be,
                      bf16_t* C) {
    dim3 g(N / 128, BT / 128);
    k_gemm<1><<<g, 256, 0, stream>>>(A, K, W, N, b0, nullptr, al, ga, be, nullptr, C);
  };
  auto scan = [&](const float* gx, const bf16_t* whh_, const float* thr,
                  const float* isyn, const float* imem, bf16_t* spk_,
                  float* osyn, float* omem, int old_, int ooff, int Hc) {
    k_zero_u32<<<1, 1, 0, stream>>>(bar);
    k_lstm_scan<<<dim3(Hc / 128), 256, (size_t)16 * Hc * 2, stream>>>(
        gx, whh_, thr, isyn, imem, spk_, osyn, omem, old_, ooff, membuf, bar, Hc, T);
  };

  // ---- embeddings + DyT (fwd, flipped, tgt) ----
  k_embed_dyt<<<BT, 256, 0, stream>>>(src, emb, nalpha, ngamma, nbeta, actA, actC, T, E);
  k_embed_dyt<<<BT, 256, 0, stream>>>(tgt, emb, nalpha, ngamma, nbeta, tgtbuf, nullptr, T, E);

  // ---- encoder: 2 layers x {fwd, bwd}; layer-1 finals -> dec init halves ----
  bf16_t* fin = actA; bf16_t* fout = actB;
  bf16_t* bin = actC; bf16_t* bout = actD;
  for (int l = 0; l < 2; ++l) {
    // forward direction
    cvt(encf[l].W_ih, wih, (long)(4 * H) * 512);
    gemm_f32(fin, 512, wih, 4 * H, encf[l].b_ih, encf[l].b_hh, gates);
    cvt(encf[l].W_hh, whh, (long)(4 * H) * H);
    if (l == 0) scan(gates, whh, encf[l].thr, nullptr, nullptr, spkbuf, ssyn, smem, H, 0, H);
    else        scan(gates, whh, encf[l].thr, nullptr, nullptr, spkbuf, dsyn, dmem, HD, 0, H);
    cvt(encf[l].fc_W, wih, (long)H * H);
    gemm_dyt(spkbuf, H, wih, H, encf[l].fc_b, encf[l].alpha, encf[l].gamma, encf[l].beta, fout);
    { bf16_t* t2 = fin; fin = fout; fout = t2; }
    // backward direction (operates on pre-flipped sequence; finals are what matter)
    cvt(encb[l].W_ih, wih, (long)(4 * H) * 512);
    gemm_f32(bin, 512, wih, 4 * H, encb[l].b_ih, encb[l].b_hh, gates);
    cvt(encb[l].W_hh, whh, (long)(4 * H) * H);
    if (l == 0) scan(gates, whh, encb[l].thr, nullptr, nullptr, spkbuf, ssyn, smem, H, 0, H);
    else        scan(gates, whh, encb[l].thr, nullptr, nullptr, spkbuf, dsyn, dmem, HD, H, H);
    cvt(encb[l].fc_W, wih, (long)H * H);
    gemm_dyt(spkbuf, H, wih, H, encb[l].fc_b, encb[l].alpha, encb[l].gamma, encb[l].beta, bout);
    { bf16_t* t2 = bin; bin = bout; bout = t2; }
  }

  // ---- decoder layer 0 (H=1024, init = concat(enc_f, enc_b) finals) ----
  cvt(dec0.W_ih, wih, (long)4096 * 512);
  gemm_f32(tgtbuf, 512, wih, 4096, dec0.b_ih, dec0.b_hh, gates);
  cvt(dec0.W_hh, whh, (long)4096 * 1024);
  scan(gates, whh, dec0.thr, dsyn, dmem, spkbuf, ssyn, smem, HD, 0, HD);
  cvt(dec0.fc_W, wih, (long)1024 * 1024);
  gemm_dyt(spkbuf, HD, wih, HD, dec0.fc_b, dec0.alpha, dec0.gamma, dec0.beta, decA);

  // ---- decoder layer 1 (zero init) ----
  cvt(dec1p.W_ih, wih, (long)4096 * 1024);
  gemm_f32(decA, 1024, wih, 4096, dec1p.b_ih, dec1p.b_hh, gates);
  cvt(dec1p.W_hh, whh, (long)4096 * 1024);
  scan(gates, whh, dec1p.thr, nullptr, nullptr, spkbuf, ssyn, smem, HD, 0, HD);
  cvt(dec1p.fc_W, wih, (long)1024 * 1024);
  gemm_dyt(spkbuf, HD, wih, HD, dec1p.fc_b, dec1p.alpha, dec1p.gamma, dec1p.beta, decB);

  // ---- final vocab projection -> d_out (fp32, [B,1,T,V] == rows b*T+t) ----
  cvt(fcW, wbig, (long)V * HD);
  gemm_f32(decB, HD, wbig, V, fcb, nullptr, (float*)d_out);
}
